// VectorQuantizer_65377992180050
// MI455X (gfx1250) — matrix operand order repaired and checked by reading, not compiled, not run on hardware
//
#include <hip/hip_runtime.h>
#include <hip/hip_bf16.h>

// ---------------------------------------------------------------------------
// VQ-VAE quantizer for gfx1250 (MI455X), bf16x3-split WMMA argmin.
//   d_in[0] = z_e   [65536 x 256] f32
//   d_in[1] = emb_w [ 2048 x 256] f32
//   d_out   = [ z_q_st (65536*256) | vq_loss (1) | indices-as-float (65536) ]
// ---------------------------------------------------------------------------

#define N_ROWS  65536
#define DIM     256
#define K_CODES 2048
#define CHUNKS  128   // K_CODES / 16 columns per WMMA tile
#define STEPS   8     // DIM / 32 (K-depth per bf16 WMMA)

typedef __bf16          v16bf __attribute__((ext_vector_type(16)));
typedef unsigned short  v16u  __attribute__((ext_vector_type(16)));
typedef float           v8f   __attribute__((ext_vector_type(8)));

// Split f32 into bf16 hi (truncated) + bf16 lo (residual). hi+lo carries ~16
// mantissa bits -> cross-products at fp32-class accuracy via 3 WMMA terms.
__device__ __forceinline__ void bf16_split(float x, unsigned short& hi, unsigned short& lo) {
    unsigned int b = __float_as_uint(x);
    hi = (unsigned short)(b >> 16);
    float r = x - __uint_as_float(b & 0xFFFF0000u);
    lo = (unsigned short)(__float_as_uint(r) >> 16);
}

// ---------------------------------------------------------------------------
// Kernel 1: e_sq[k] = sum_d emb[k][d]^2   (z_sq is row-constant -> dropped)
// ---------------------------------------------------------------------------
__global__ __launch_bounds__(256) void vq_esq_kernel(const float* __restrict__ emb,
                                                     float* __restrict__ e_sq) {
    int k = blockIdx.x * 256 + threadIdx.x;
    if (k >= K_CODES) return;
    const float* row = emb + (size_t)k * DIM;
    float s = 0.f;
    for (int d = 0; d < DIM; ++d) { float v = row[d]; s += v * v; }
    e_sq[k] = s;
}

// ---------------------------------------------------------------------------
// Kernel 2: pre-swizzle emb into bf16 hi/lo B-fragment order.
// Slot t = (chunk c, step s, lane L); lane L holds column n = c*16 + (L&15),
// dims koff..koff+15 with koff = 32*s + (L<16 ? 0 : 16)  (contiguous k order, same as A).
// ---------------------------------------------------------------------------
__global__ __launch_bounds__(256) void vq_embswz_kernel(const float* __restrict__ emb,
                                                        unsigned short* __restrict__ ehi,
                                                        unsigned short* __restrict__ elo) {
    int t = blockIdx.x * 256 + threadIdx.x;   // 0 .. CHUNKS*STEPS*32-1
    int c = t >> 8;        // chunk (256 slots per chunk)
    int r = t & 255;
    int s = r >> 5;        // step
    int L = r & 31;        // lane slot
    int code = c * 16 + (L & 15);
    int koff = 32 * s + ((L < 16) ? 0 : 16);
    const float* src = emb + (size_t)code * DIM + koff;
    size_t base = (size_t)t * 16;
#pragma unroll
    for (int j = 0; j < 16; ++j) {
        unsigned short h, l;
        bf16_split(src[j], h, l);
        ehi[base + j] = h;
        elo[base + j] = l;
    }
}

// ---------------------------------------------------------------------------
// Kernel 3: main argmin. 1 wave = 16 rows of z; 8 waves/block = 128 rows.
// Loops 128 chunks of 16 codes; emb chunk staged in LDS (read once per block).
// dist(n,k) = e_sq[k] - 2*cross(n,k)   (argmin-equivalent)
// ---------------------------------------------------------------------------
__global__ __launch_bounds__(256) void vq_argmin_kernel(const float* __restrict__ z_e,
                                                        const unsigned short* __restrict__ ehi,
                                                        const unsigned short* __restrict__ elo,
                                                        const float* __restrict__ e_sq,
                                                        int* __restrict__ idx_out) {
    __shared__ unsigned short sb_hi[STEPS * 32 * 16];   // 8 KB
    __shared__ unsigned short sb_lo[STEPS * 32 * 16];   // 8 KB

    const int tid  = threadIdx.x;
    const int wave = tid >> 5;
    const int lane = tid & 31;
    const int rowBase = (blockIdx.x * 8 + wave) * 16;
    const int half = (lane < 16) ? 0 : 16;   // contiguous half-wave K offset (matches B)
    const float* zrow = z_e + (size_t)(rowBase + (lane & 15)) * DIM;

    // Build A fragments (hi/lo) in registers, once per wave.
    // Lane L, step s: elems 0..15 = K 32s+half+0..15
    v16u a_hi[STEPS], a_lo[STEPS];
#pragma unroll
    for (int s = 0; s < STEPS; ++s) {
#pragma unroll
        for (int j = 0; j < 8; ++j) {
            unsigned short h, l;
            bf16_split(zrow[32 * s + half + j], h, l);
            a_hi[s][j] = h;  a_lo[s][j] = l;
            bf16_split(zrow[32 * s + half + 8 + j], h, l);
            a_hi[s][8 + j] = h;  a_lo[s][8 + j] = l;
        }
    }

    float rmin[8];
    int   ridx[8];
#pragma unroll
    for (int v = 0; v < 8; ++v) { rmin[v] = __builtin_huge_valf(); ridx[v] = 0; }

    const v16u* g_hi = (const v16u*)ehi;
    const v16u* g_lo = (const v16u*)elo;
    v16u* s_hi = (v16u*)sb_hi;
    v16u* s_lo = (v16u*)sb_lo;

#pragma unroll 1
    for (int c = 0; c < CHUNKS; ++c) {
        __syncthreads();                       // protect LDS from previous compute
        s_hi[tid] = g_hi[c * 256 + tid];       // 8 KB coalesced stage (identity layout)
        s_lo[tid] = g_lo[c * 256 + tid];
        __syncthreads();

        v8f acc0 = {};  // z_hi * e_hi
        v8f acc1 = {};  // z_lo * e_hi
        v8f acc2 = {};  // z_hi * e_lo   (3 independent chains -> ILP in matrix pipe)
#pragma unroll
        for (int s = 0; s < STEPS; ++s) {
            v16bf ah = __builtin_bit_cast(v16bf, a_hi[s]);
            v16bf al = __builtin_bit_cast(v16bf, a_lo[s]);
            v16bf bh = __builtin_bit_cast(v16bf, s_hi[s * 32 + lane]);
            v16bf bl = __builtin_bit_cast(v16bf, s_lo[s * 32 + lane]);
            acc0 = __builtin_amdgcn_wmma_f32_16x16x32_bf16(false, ah, false, bh, (short)0, acc0, false, false);
            acc1 = __builtin_amdgcn_wmma_f32_16x16x32_bf16(false, al, false, bh, (short)0, acc1, false, false);
            acc2 = __builtin_amdgcn_wmma_f32_16x16x32_bf16(false, ah, false, bl, (short)0, acc2, false, false);
        }

        // C layout: slot v holds (row = rowBase + v + 8*(lane/16), col = c*16 + lane%16)
        int col = c * 16 + (lane & 15);
        float esq = e_sq[col];
#pragma unroll
        for (int v = 0; v < 8; ++v) {
            float cross = acc0[v] + acc1[v] + acc2[v];
            float d = esq - 2.0f * cross;
            if (d < rmin[v]) { rmin[v] = d; ridx[v] = col; }   // strict < keeps first (lowest) index within lane
        }
    }

    // Reduce (min, idx) across the 16 lanes of each half-wave; ties -> lower index.
#pragma unroll
    for (int v = 0; v < 8; ++v) {
        float m = rmin[v];
        int   ix = ridx[v];
#pragma unroll
        for (int off = 8; off >= 1; off >>= 1) {
            float m2 = __shfl_xor(m, off, 32);
            int   i2 = __shfl_xor(ix, off, 32);
            if (m2 < m || (m2 == m && i2 < ix)) { m = m2; ix = i2; }
        }
        if ((lane & 15) == 0)
            idx_out[rowBase + v + ((lane < 16) ? 0 : 8)] = ix;
    }
}

// ---------------------------------------------------------------------------
// Kernel 4: gather z_q (= forward value of z_q_st), per-row SSE partials,
// indices as float. One block per row.
// ---------------------------------------------------------------------------
__global__ __launch_bounds__(256) void vq_gather_kernel(const float* __restrict__ z_e,
                                                        const float* __restrict__ emb,
                                                        const int* __restrict__ idx_in,
                                                        float* __restrict__ out_zq,
                                                        float* __restrict__ out_idx,
                                                        float* __restrict__ partials) {
    __shared__ float sred[256];
    int n = blockIdx.x;
    int t = threadIdx.x;
    int k = idx_in[n];
    float e = emb[(size_t)k * DIM + t];
    float z = z_e[(size_t)n * DIM + t];
    out_zq[(size_t)n * DIM + t] = e;
    float d = e - z;
    sred[t] = d * d;
    __syncthreads();
    for (int off = 128; off >= 1; off >>= 1) {
        if (t < off) sred[t] += sred[t + off];
        __syncthreads();
    }
    if (t == 0) {
        partials[n] = sred[0];
        out_idx[n] = (float)k;
    }
}

// ---------------------------------------------------------------------------
// Kernel 5: deterministic final loss reduction.
// vq_loss = (1 + 0.25) * mean((z_q - z_e)^2)   (both loss terms equal forward)
// ---------------------------------------------------------------------------
__global__ __launch_bounds__(256) void vq_loss_kernel(const float* __restrict__ partials,
                                                      float* __restrict__ out_loss) {
    __shared__ float sred[256];
    int t = threadIdx.x;
    float s = 0.f;
    for (int i = t; i < N_ROWS; i += 256) s += partials[i];
    sred[t] = s;
    __syncthreads();
    for (int off = 128; off >= 1; off >>= 1) {
        if (t < off) sred[t] += sred[t + off];
        __syncthreads();
    }
    if (t == 0)
        out_loss[0] = 1.25f * sred[0] / ((float)N_ROWS * (float)DIM);
}

// ---------------------------------------------------------------------------
extern "C" void kernel_launch(void* const* d_in, const int* in_sizes, int n_in,
                              void* d_out, int out_size, void* d_ws, size_t ws_size,
                              hipStream_t stream) {
    const float* z_e = (const float*)d_in[0];
    const float* emb = (const float*)d_in[1];

    float* out      = (float*)d_out;
    float* out_zq   = out;                                   // 65536*256
    float* out_loss = out + (size_t)N_ROWS * DIM;            // 1
    float* out_idx  = out_loss + 1;                          // 65536

    // Workspace carve (~2.52 MB total), all 256B-aligned offsets:
    char* ws = (char*)d_ws;
    unsigned short* ehi      = (unsigned short*)(ws);                         // 1 MB
    unsigned short* elo      = (unsigned short*)(ws + (1u << 20));            // 1 MB
    float*          e_sq     = (float*)(ws + (2u << 20));                     // 8 KB
    int*            idx      = (int*)  (ws + (2u << 20) + 8192);              // 256 KB
    float*          partials = (float*)(ws + (2u << 20) + 8192 + 262144);     // 256 KB

    vq_esq_kernel   <<<K_CODES / 256, 256, 0, stream>>>(emb, e_sq);
    vq_embswz_kernel<<<(CHUNKS * STEPS * 32) / 256, 256, 0, stream>>>(emb, ehi, elo);
    vq_argmin_kernel<<<N_ROWS / 128, 256, 0, stream>>>(z_e, ehi, elo, e_sq, idx);
    vq_gather_kernel<<<N_ROWS, 256, 0, stream>>>(z_e, emb, idx, out_zq, out_idx, partials);
    vq_loss_kernel  <<<1, 256, 0, stream>>>(partials, out_loss);
}
